// TNModule_63393717289327
// MI455X (gfx1250) — matrix operand order repaired and checked by reading, not compiled, not run on hardware
//
#include <hip/hip_runtime.h>
#include <hip/hip_bf16.h>
#include <stdint.h>

#define NT       2560      // NUM_NODES + SEQ_LEN
#define LATENT   32
#define NBATCH   4
#define SELF_EPS 0.5f
#define WAVES    4

typedef __attribute__((ext_vector_type(16))) __bf16   v16bf;
typedef __attribute__((ext_vector_type(8)))  float    v8f;
typedef __attribute__((ext_vector_type(4)))  uint32_t uv4;
typedef __attribute__((ext_vector_type(4)))  float    fv4;

union ABf { v16bf v; uv4 q[2]; uint32_t w[8]; };

// compiler-only ordering fence (LDS u16 stores vs u32-typed reloads)
#define MEM_FENCE() asm volatile("" ::: "memory")

static __device__ __forceinline__ uint32_t cvt2bf(float a, float b) {
  uint32_t ua = __float_as_uint(a), ub = __float_as_uint(b);
  ua = (ua + 0x7FFFu + ((ua >> 16) & 1u)) >> 16;
  ub = (ub + 0x7FFFu + ((ub >> 16) & 1u)) >> 16;
  return ua | (ub << 16);
}
static __device__ __forceinline__ uint16_t cvt1bf(float a) {
  uint32_t ua = __float_as_uint(a);
  return (uint16_t)((ua + 0x7FFFu + ((ua >> 16) & 1u)) >> 16);
}
static __device__ __forceinline__ float dtanh_pos(float x) {
#if defined(__has_builtin)
#if __has_builtin(__builtin_amdgcn_tanhf)
  return __builtin_amdgcn_tanhf(x);   // v_tanh_f32 on gfx1250
#else
  return tanhf(x);
#endif
#else
  return tanhf(x);
#endif
}

// -------- prep: f32 X -> bf16 row-major + bf16 transposed ----------
__global__ void prep_bf16(const float* __restrict__ X,
                          uint16_t* __restrict__ Xbf,
                          uint16_t* __restrict__ XbfT) {
  int i = blockIdx.x * blockDim.x + threadIdx.x;
  if (i >= NBATCH * NT) return;
  int b = i / NT, n = i % NT;
  const float* xr = X + (size_t)i * LATENT;
  uint32_t d[16];
#pragma unroll
  for (int j = 0; j < 16; j++) d[j] = cvt2bf(xr[2 * j], xr[2 * j + 1]);
  uv4* dst = (uv4*)(Xbf + (size_t)i * LATENT);
#pragma unroll
  for (int j = 0; j < 4; j++) {
    uv4 q = {d[4 * j], d[4 * j + 1], d[4 * j + 2], d[4 * j + 3]};
    dst[j] = q;
  }
  uint16_t* t = XbfT + (size_t)b * LATENT * NT + n;
#pragma unroll
  for (int f = 0; f < LATENT; f++) t[(size_t)f * NT] = cvt1bf(xr[f]);
}

// -------- fused layer: h' = elu(h Ws + (f(X X^T) h) Wn + b) --------
// Tiles computed in transposed orientation: lane = n, VGPR = feature.
template <int MODE>
__global__ __launch_bounds__(128) void gsage_layer(
    const uint16_t* __restrict__ hrow,  // [B][NT][32] bf16 layer input
    const uint16_t* __restrict__ hT,    // [B][32][NT] bf16 layer input (transposed)
    const uint16_t* __restrict__ Xrow,  // [B][NT][32] bf16 (adjacency operands)
    const float* __restrict__ Ws, const float* __restrict__ Wn,
    const float* __restrict__ bias,
    uint16_t* __restrict__ orow,        // MODE==0 outputs
    uint16_t* __restrict__ oT,
    float* __restrict__ ofin)           // MODE==1 output
{
  __shared__ __align__(16) uint16_t wsT16[32 * 32];
  __shared__ __align__(16) uint16_t wnT16[32 * 32];
  __shared__ float bias_s[32];
  __shared__ __align__(16) uint16_t s16[WAVES][16 * 32];   // per-wave S tile [n][m] bf16
  __shared__ __align__(16) float aggf[WAVES][2][32][8];    // partial agg reduction
  __shared__ __align__(16) uint16_t agg16[16 * 32];        // agg [n][d] bf16
  __shared__ __align__(16) uint16_t ot16[32 * 16];         // out^T staging

  const int tid = threadIdx.x;
  const int wave = tid >> 5, lane = tid & 31;
  const int half = lane >> 4, l16 = lane & 15;
  const int b = blockIdx.x / (NT / 16);
  const int n0 = (blockIdx.x % (NT / 16)) * 16;

  // stage transposed weights (bf16) + bias
  for (int i = tid; i < 32 * 32; i += blockDim.x) {
    int dr = i >> 5, dc = i & 31;
    wsT16[dc * 32 + dr] = cvt1bf(Ws[i]);
    wnT16[dc * 32 + dr] = cvt1bf(Wn[i]);
  }
  if (tid < 32) bias_s[tid] = bias[tid];
  __syncthreads();

  const size_t rowbase = (size_t)b * NT * 32;  // u16 elements
  const size_t tbase   = (size_t)b * 32 * NT;

  // A operand: X rows n0..n0+15 (loop invariant)
  ABf aX;
  {
    const char* p = (const char*)(Xrow + rowbase + (size_t)(n0 + l16) * 32);
    aX.q[0] = *(const uv4*)(p + 16 * half);
    aX.q[1] = *(const uv4*)(p + 32 + 16 * half);
  }

  // operand loader for one 32-column chunk
  auto load_chunk = [&](int m, ABf& b0, ABf& b1, ABf& A0, ABf& A1) {
    const char* p0 = (const char*)(Xrow + rowbase + (size_t)(m + l16) * 32);
    b0.q[0] = *(const uv4*)(p0 + 16 * half);
    b0.q[1] = *(const uv4*)(p0 + 32 + 16 * half);
    const char* p1 = (const char*)(Xrow + rowbase + (size_t)(m + 16 + l16) * 32);
    b1.q[0] = *(const uv4*)(p1 + 16 * half);
    b1.q[1] = *(const uv4*)(p1 + 32 + 16 * half);
    const char* pa = (const char*)(hT + tbase + (size_t)l16 * NT + m);
    A0.q[0] = *(const uv4*)(pa + 16 * half);
    A0.q[1] = *(const uv4*)(pa + 32 + 16 * half);
    const char* pb = (const char*)(hT + tbase + (size_t)(16 + l16) * NT + m);
    A1.q[0] = *(const uv4*)(pb + 16 * half);
    A1.q[1] = *(const uv4*)(pb + 32 + 16 * half);
  };

  v8f acc0 = {}, acc1 = {};   // agg^T tiles: (d 0-15 x n), (d 16-31 x n)
  uint16_t* sw = s16[wave];

  ABf bx0, bx1, a0, a1;
  int mc = wave * 32;                    // always < NT (128 <= NT)
  load_chunk(mc, bx0, bx1, a0, a1);

  while (mc < NT) {
    const int mn = mc + 32 * WAVES;
    // software pipeline: issue next chunk's loads before computing this one
    ABf nb0, nb1, na0, na1;
    if (mn < NT) load_chunk(mn, nb0, nb1, na0, na1);

    // ---- adjacency tiles: S = X[n0:,:] @ X[mc:,:]^T  (two 16x16) ----
    v8f cz = {};
    v8f s0 = __builtin_amdgcn_wmma_f32_16x16x32_bf16(false, aX.v, false, bx0.v, (short)0, cz, false, false);
    v8f s1 = __builtin_amdgcn_wmma_f32_16x16x32_bf16(false, aX.v, false, bx1.v, (short)0, cz, false, false);

    // ---- tanh(relu(.)) (+self-loop on the one diagonal chunk) ----
    const bool diag = (mc < n0 + 16) && (mc + 32 > n0);   // wave-uniform
#pragma unroll
    for (int r = 0; r < 8; r++) {
      int nl = r + 8 * half;
      {
        float x = s0[r];
        float y = x > 0.f ? dtanh_pos(x) : 0.f;
        if (diag && (n0 + nl == mc + l16)) y += SELF_EPS;
        sw[nl * 32 + l16] = cvt1bf(y);
      }
      {
        float x = s1[r];
        float y = x > 0.f ? dtanh_pos(x) : 0.f;
        if (diag && (n0 + nl == mc + 16 + l16)) y += SELF_EPS;
        sw[nl * 32 + 16 + l16] = cvt1bf(y);
      }
    }
    MEM_FENCE();   // keep u16 S stores ordered before u32 reloads

    // ---- agg^T += hT[:, mc:mc+32] @ S^T ----
    ABf bS;
    bS.q[0] = *(const uv4*)((const char*)sw + l16 * 64 + 16 * half);
    bS.q[1] = *(const uv4*)((const char*)sw + l16 * 64 + 32 + 16 * half);
    acc0 = __builtin_amdgcn_wmma_f32_16x16x32_bf16(false, a0.v, false, bS.v, (short)0, acc0, false, false);
    acc1 = __builtin_amdgcn_wmma_f32_16x16x32_bf16(false, a1.v, false, bS.v, (short)0, acc1, false, false);
    MEM_FENCE();   // next iteration overwrites sw: keep reads before stores

    bx0 = nb0; bx1 = nb1; a0 = na0; a1 = na1;
    mc = mn;
  }

  // ---- cross-wave reduction of partial agg ----
#pragma unroll
  for (int r = 0; r < 8; r++) {
    aggf[wave][0][lane][r] = acc0[r];
    aggf[wave][1][lane][r] = acc1[r];
  }
  __syncthreads();

  if (wave == 0) {
#pragma unroll
    for (int r = 0; r < 8; r++) {
      float a0v = 0.f, a1v = 0.f;
#pragma unroll
      for (int w = 0; w < WAVES; w++) { a0v += aggf[w][0][lane][r]; a1v += aggf[w][1][lane][r]; }
      acc0[r] = a0v; acc1[r] = a1v;
    }
    // agg -> LDS [n][d] bf16 (contiguous d runs per lane)
    {
      uint32_t pk[4];
#pragma unroll
      for (int j = 0; j < 4; j++) pk[j] = cvt2bf(acc0[2 * j], acc0[2 * j + 1]);
      uv4 q0 = {pk[0], pk[1], pk[2], pk[3]};
      *(uv4*)((char*)agg16 + l16 * 64 + 16 * half) = q0;
#pragma unroll
      for (int j = 0; j < 4; j++) pk[j] = cvt2bf(acc1[2 * j], acc1[2 * j + 1]);
      uv4 q1 = {pk[0], pk[1], pk[2], pk[3]};
      *(uv4*)((char*)agg16 + l16 * 64 + 32 + 16 * half) = q1;
    }
    MEM_FENCE();

    // ---- out^T = Ws^T h^T + Wn^T agg^T ----
    ABf bh, bagg;
    {
      const char* p = (const char*)(hrow + rowbase + (size_t)(n0 + l16) * 32);
      bh.q[0] = *(const uv4*)(p + 16 * half);
      bh.q[1] = *(const uv4*)(p + 32 + 16 * half);
    }
    bagg.q[0] = *(const uv4*)((const char*)agg16 + l16 * 64 + 16 * half);
    bagg.q[1] = *(const uv4*)((const char*)agg16 + l16 * 64 + 32 + 16 * half);

    ABf aws0, aws1, awn0, awn1;
    {
      const char* p = (const char*)wsT16;
      aws0.q[0] = *(const uv4*)(p + l16 * 64 + 16 * half);
      aws0.q[1] = *(const uv4*)(p + l16 * 64 + 32 + 16 * half);
      aws1.q[0] = *(const uv4*)(p + (16 + l16) * 64 + 16 * half);
      aws1.q[1] = *(const uv4*)(p + (16 + l16) * 64 + 32 + 16 * half);
      const char* p2 = (const char*)wnT16;
      awn0.q[0] = *(const uv4*)(p2 + l16 * 64 + 16 * half);
      awn0.q[1] = *(const uv4*)(p2 + l16 * 64 + 32 + 16 * half);
      awn1.q[0] = *(const uv4*)(p2 + (16 + l16) * 64 + 16 * half);
      awn1.q[1] = *(const uv4*)(p2 + (16 + l16) * 64 + 32 + 16 * half);
    }
    v8f cz = {};
    v8f o0 = __builtin_amdgcn_wmma_f32_16x16x32_bf16(false, aws0.v, false, bh.v, (short)0, cz, false, false);
    o0 = __builtin_amdgcn_wmma_f32_16x16x32_bf16(false, awn0.v, false, bagg.v, (short)0, o0, false, false);
    v8f o1 = __builtin_amdgcn_wmma_f32_16x16x32_bf16(false, aws1.v, false, bh.v, (short)0, cz, false, false);
    o1 = __builtin_amdgcn_wmma_f32_16x16x32_bf16(false, awn1.v, false, bagg.v, (short)0, o1, false, false);

    float y0[8], y1[8];
#pragma unroll
    for (int r = 0; r < 8; r++) {
      float x0 = o0[r] + bias_s[8 * half + r];
      float x1 = o1[r] + bias_s[16 + 8 * half + r];
      y0[r] = x0 > 0.f ? x0 : (__expf(x0) - 1.f);
      y1[r] = x1 > 0.f ? x1 : (__expf(x1) - 1.f);
    }

    int gn = n0 + l16;
    if (MODE == 0) {
      uint32_t pk0[4], pk1[4];
#pragma unroll
      for (int j = 0; j < 4; j++) {
        pk0[j] = cvt2bf(y0[2 * j], y0[2 * j + 1]);
        pk1[j] = cvt2bf(y1[2 * j], y1[2 * j + 1]);
      }
      uv4 q0 = {pk0[0], pk0[1], pk0[2], pk0[3]};
      uv4 q1 = {pk1[0], pk1[1], pk1[2], pk1[3]};
      char* pr = (char*)(orow + rowbase + (size_t)gn * 32);
      *(uv4*)(pr + 16 * half) = q0;
      *(uv4*)(pr + 32 + 16 * half) = q1;
      // transposed output via LDS for coalesced stores
#pragma unroll
      for (int r = 0; r < 8; r++) {
        ot16[(8 * half + r) * 16 + l16]      = (uint16_t)(pk0[r >> 1] >> (16 * (r & 1)));
        ot16[(16 + 8 * half + r) * 16 + l16] = (uint16_t)(pk1[r >> 1] >> (16 * (r & 1)));
      }
      MEM_FENCE();
      {
        int dp = lane;  // row of ot16 (feature)
        uv4 r0 = *(const uv4*)((const char*)ot16 + dp * 32);
        uv4 r1 = *(const uv4*)((const char*)ot16 + dp * 32 + 16);
        char* pt = (char*)(oT + tbase + (size_t)dp * NT + n0);
        *(uv4*)pt = r0;
        *(uv4*)(pt + 16) = r1;
      }
    } else {
      float* po = ofin + ((size_t)b * NT + gn) * 32;
      fv4 f0a = {y0[0], y0[1], y0[2], y0[3]};
      fv4 f0b = {y0[4], y0[5], y0[6], y0[7]};
      fv4 f1a = {y1[0], y1[1], y1[2], y1[3]};
      fv4 f1b = {y1[4], y1[5], y1[6], y1[7]};
      *(fv4*)(po + 8 * half)          = f0a;
      *(fv4*)(po + 8 * half + 4)      = f0b;
      *(fv4*)(po + 16 + 8 * half)     = f1a;
      *(fv4*)(po + 16 + 8 * half + 4) = f1b;
    }
  }
}

extern "C" void kernel_launch(void* const* d_in, const int* in_sizes, int n_in,
                              void* d_out, int out_size, void* d_ws, size_t ws_size,
                              hipStream_t stream) {
  const float* X  = (const float*)d_in[0];
  const float* Ws = (const float*)d_in[1];
  const float* Wn = (const float*)d_in[2];
  const float* bb = (const float*)d_in[3];

  uint16_t* wsb   = (uint16_t*)d_ws;
  uint16_t* Xbf   = wsb;
  uint16_t* XbfT  = Xbf  + (size_t)NBATCH * NT * LATENT;
  uint16_t* h1bf  = XbfT + (size_t)NBATCH * NT * LATENT;
  uint16_t* h1bfT = h1bf + (size_t)NBATCH * NT * LATENT;

  prep_bf16<<<(NBATCH * NT + 255) / 256, 256, 0, stream>>>(X, Xbf, XbfT);

  dim3 grid(NBATCH * (NT / 16));
  // layer 0: h = X
  gsage_layer<0><<<grid, 128, 0, stream>>>(Xbf, XbfT, Xbf, Ws, Wn, bb,
                                           h1bf, h1bfT, nullptr);
  // layer 1: h = h1, adjacency recomputed from X, write f32 output
  gsage_layer<1><<<grid, 128, 0, stream>>>(h1bf, h1bfT, Xbf,
                                           Ws + 32 * 32, Wn + 32 * 32, bb + 32,
                                           nullptr, nullptr, (float*)d_out);
}
